// LightMLPInterpolate_55637006352992
// MI455X (gfx1250) — compile-verified, matrix operand order checked by reading, not run on hardware
//
#include <hip/hip_runtime.h>
#include <hip/hip_bf16.h>

typedef __attribute__((ext_vector_type(16))) _Float16 v16h;
typedef __attribute__((ext_vector_type(8)))  _Float16 v8h;
typedef __attribute__((ext_vector_type(8)))  float    v8f;
typedef __attribute__((ext_vector_type(4)))  int      i32x4;

union V16 { v16h v; v8h h[2]; };

#define HOUT 512
#define WIN  128
#define CCH  64
#define NPIX (HOUT*HOUT)          // 262144
#define KCHUNKS 18                // K = 9 taps * 64 ch = 576 = 18 * 32
#define LDS_COLS 34               // 32-pixel strip + halo
#define LDS_HALVES (3*LDS_COLS*CCH)   // 6528 halves = 13056 B
#define STAGE_VECS (3*LDS_COLS*8)     // 816 x 16B vectors per tile

// ---- CDNA5 async global->LDS copy (ASYNCcnt) with safe fallback -----------
#if defined(__has_builtin)
#if __has_builtin(__builtin_amdgcn_global_load_async_to_lds_b128)
#define HAVE_ASYNC_LDS 1
#endif
#if __has_builtin(__builtin_amdgcn_s_wait_asynccnt)
#define HAVE_WAIT_ASYNC 1
#endif
#endif

__device__ __forceinline__ void async_copy16(_Float16* dst, const _Float16* src)
{
#ifdef HAVE_ASYNC_LDS
    // prototype: (int4 AS1* gsrc, int4 AS3* ldst, imm offset, imm cpol)
    i32x4* s_generic = (i32x4*)const_cast<_Float16*>(src);
    i32x4* d_generic = (i32x4*)dst;
    __builtin_amdgcn_global_load_async_to_lds_b128(
        (__attribute__((address_space(1))) i32x4*)s_generic,
        (__attribute__((address_space(3))) i32x4*)d_generic,
        0, 0);
#else
    *(uint4*)dst = *(const uint4*)src;
#endif
}

__device__ __forceinline__ void wait_async0()
{
#ifdef HAVE_ASYNC_LDS
#ifdef HAVE_WAIT_ASYNC
    __builtin_amdgcn_s_wait_asynccnt(0);
#else
    asm volatile("s_wait_asynccnt 0" ::: "memory");
#endif
#endif
}

// ---------------------------------------------------------------------------
// Kernel A: per-output-pixel MLP weights + 3x3 gather -> feat [512][512][64] f16
// ---------------------------------------------------------------------------
__global__ __launch_bounds__(256) void kA_feat(
    const float* __restrict__ x,      // [64][128][128]
    const float* __restrict__ w1,     // [3][64]
    const float* __restrict__ b1,     // [64]
    const float* __restrict__ w2,     // [64][9]
    const float* __restrict__ b2,     // [9]
    _Float16* __restrict__ feat)      // [512][512][64]
{
    int p  = blockIdx.x * 256 + threadIdx.x;
    int oy = p >> 9;
    int ox = p & 511;

    float oyc = -1.0f + (2.0f * (float)oy + 1.0f) * (1.0f / 512.0f);
    float oxc = -1.0f + (2.0f * (float)ox + 1.0f) * (1.0f / 512.0f);
    // nearest src index, round half-to-even like jnp.round
    float fy = rintf(((oyc + 1.0f) * 128.0f - 1.0f) * 0.5f);
    float fx = rintf(((oxc + 1.0f) * 128.0f - 1.0f) * 0.5f);
    int iy = (int)fy; iy = iy < 0 ? 0 : (iy > 127 ? 127 : iy);
    int ix = (int)fx; ix = ix < 0 ? 0 : (ix > 127 ? 127 : ix);
    float qy = -1.0f + (2.0f * (float)iy + 1.0f) * (1.0f / 128.0f);
    float qx = -1.0f + (2.0f * (float)ix + 1.0f) * (1.0f / 128.0f);
    float rel_y = (oyc - qy) * 128.0f;
    float rel_x = (oxc - qx) * 128.0f;
    const float scale = 0.25f;   // 128/512

    float logit[9];
#pragma unroll
    for (int j = 0; j < 9; ++j) logit[j] = b2[j];
#pragma unroll 4
    for (int k = 0; k < 64; ++k) {
        float h = rel_y * w1[k] + rel_x * w1[64 + k] + scale * w1[128 + k] + b1[k];
        h = fmaxf(h, 0.0f);
#pragma unroll
        for (int j = 0; j < 9; ++j) logit[j] += h * w2[k * 9 + j];
    }
    float mx = logit[0];
#pragma unroll
    for (int j = 1; j < 9; ++j) mx = fmaxf(mx, logit[j]);
    float wt[9]; float s = 0.0f;
#pragma unroll
    for (int j = 0; j < 9; ++j) { wt[j] = __expf(logit[j] - mx); s += wt[j]; }
    float inv = 1.0f / s;
#pragma unroll
    for (int j = 0; j < 9; ++j) wt[j] *= inv;

    float acc[64];
#pragma unroll
    for (int c = 0; c < 64; ++c) acc[c] = 0.0f;

#pragma unroll
    for (int ky = 0; ky < 3; ++ky) {
#pragma unroll
        for (int kx = 0; kx < 3; ++kx) {
            int r  = iy + ky - 1;
            int cl = ix + kx - 1;
            if (r >= 0 && r < 128 && cl >= 0 && cl < 128) {
                float w = wt[ky * 3 + kx];
                const float* xp = x + r * 128 + cl;
#pragma unroll
                for (int c = 0; c < 64; ++c) acc[c] += w * xp[c * (128 * 128)];
            }
        }
    }

    _Float16* fp = feat + ((size_t)p << 6);
#pragma unroll
    for (int q = 0; q < 8; ++q) {
        v8h v;
#pragma unroll
        for (int t = 0; t < 8; ++t) v[t] = (_Float16)acc[q * 8 + t];
        *(v8h*)(fp + q * 8) = v;
    }
}

// ---------------------------------------------------------------------------
// Kernel B0: repack conv1_w f32 [oc][ic][3][3] into per-lane WMMA B-fragment
// layout: Bpack[chunk c][g01][n][16 halves]; K = c*32 + kk,
// kk = g01*8 + (h<8 ? h : h+8)  (v0..3 hold K g..g+7, v4..7 hold K g+16..g+23)
// ---------------------------------------------------------------------------
__global__ __launch_bounds__(256) void kB0_pack(
    const float* __restrict__ w, _Float16* __restrict__ bp)
{
    int idx = blockIdx.x * 256 + threadIdx.x;
    if (idx >= KCHUNKS * 2 * 64 * 16) return;
    int h   = idx & 15;
    int n   = (idx >> 4) & 63;
    int g01 = (idx >> 10) & 1;
    int c   = idx >> 11;
    int kk  = g01 * 8 + (h < 8 ? h : h + 8);
    int tap = c >> 1;                       // ky*3+kx
    int ic  = ((c & 1) << 5) + kk;
    bp[idx] = (_Float16)w[n * 576 + ic * 9 + tap];
}

// ---------------------------------------------------------------------------
// Kernel B: conv1 3x3 64->64 + bias + ReLU as implicit GEMM with WMMA f16->f32.
// Block: 8 waves = 2(M) x 4(N); tile 32 pixels x 64 oc; 4 rows per block.
// Double-buffered LDS; next row's halo tile streamed in with
// GLOBAL_LOAD_ASYNC_TO_LDS_B128 while current row's 18 WMMAs execute.
// ---------------------------------------------------------------------------
__global__ __launch_bounds__(256) void kB_conv1(
    const _Float16* __restrict__ feat,   // [512][512][64]
    const _Float16* __restrict__ bp,     // Bpack
    const float*    __restrict__ bias,   // [64]
    _Float16*       __restrict__ outp)   // [512][512][64]
{
    __shared__ __align__(16) _Float16 sA[2][LDS_HALVES];   // 2 x 13056 B

    int tid  = threadIdx.x;
    int lane = tid & 31;
    int wave = tid >> 5;
    int wm   = wave >> 2;          // 0..1  (M tile)
    int wn   = wave & 3;           // 0..3  (N tile)
    int xb   = (blockIdx.x & 15) << 5;         // strip x0 (0,32,...,480)
    int y0   = (blockIdx.x >> 4) << 2;         // 4 rows per block
    int oc   = (wn << 4) + (lane & 15);
    int g    = (lane >= 16) ? 8 : 0;
    int g01  = lane >> 4;
    float bz = bias[oc];

    // B fragments resident in registers for all 4 rows
    V16 bf[KCHUNKS];
#pragma unroll
    for (int c = 0; c < KCHUNKS; ++c) {
        const v8h* s8 = (const v8h*)(bp + ((size_t)((c * 2 + g01) * 64 + oc) << 4));
        bf[c].h[0] = s8[0];
        bf[c].h[1] = s8[1];
    }

    // Stage 3 rows x 34 cols x 64 ch halo tile of row y into LDS buffer `buf`.
    // In-bounds 16B vectors: async global->LDS (ASYNCcnt). OOB: ds_store zeros.
    auto stage = [&](int buf, int y) {
#pragma unroll
        for (int ii = 0; ii < 4; ++ii) {
            int i = tid + ii * 256;
            if (i < STAGE_VECS) {
                int rr  = i / (LDS_COLS * 8);
                int rem = i - rr * (LDS_COLS * 8);
                int cc  = rem >> 3;
                int q   = rem & 7;
                int ry  = y + rr - 1;
                int cx  = xb + cc - 1;
                _Float16* dst = &sA[buf][((rr * LDS_COLS + cc) << 6) + (q << 3)];
                if (ry >= 0 && ry < 512 && cx >= 0 && cx < 512) {
                    const _Float16* src =
                        feat + ((((size_t)ry << 9) + cx) << 6) + (q << 3);
                    async_copy16(dst, src);
                } else {
                    *(uint4*)dst = make_uint4(0u, 0u, 0u, 0u);
                }
            }
        }
    };

    stage(0, y0);
    wait_async0();
    __syncthreads();

    for (int ri = 0; ri < 4; ++ri) {
        int y   = y0 + ri;
        int cur = ri & 1;
        // Kick off next row's tile into the other buffer (its readers passed
        // the barrier at the end of iteration ri-1), then compute this row.
        if (ri < 3) stage(cur ^ 1, y + 1);

        v8f acc;
#pragma unroll
        for (int t = 0; t < 8; ++t) acc[t] = bz;

#pragma unroll
        for (int c = 0; c < KCHUNKS; ++c) {
            const int tap = c >> 1;
            const int ky  = tap / 3;
            const int kx  = tap - ky * 3;
            const int icb = (c & 1) << 5;
            int pcol = (wm << 4) + (lane & 15) + kx;          // 0..33
            int base = ((ky * LDS_COLS + pcol) << 6) + icb + g;
            V16 a;
            a.h[0] = *(const v8h*)&sA[cur][base];        // v0..3 : K g..g+7
            a.h[1] = *(const v8h*)&sA[cur][base + 16];   // v4..7 : K g+16..g+23
            acc = __builtin_amdgcn_wmma_f32_16x16x32_f16(
                false, a.v, false, bf[c].v, (short)0, acc, false, false);
        }

        // ReLU + store: lane holds oc fixed; vgpr r -> pixel (wm*16 + r + 8*(lane>=16))
#pragma unroll
        for (int r = 0; r < 8; ++r) {
            int p = (wm << 4) + r + ((lane >= 16) ? 8 : 0);
            float v = fmaxf(acc[r], 0.0f);
            outp[((((size_t)y << 9) + xb + p) << 6) + oc] = (_Float16)v;
        }

        wait_async0();       // next buffer's async fills complete (our lanes)
        __syncthreads();     // all threads' fills + this row's LDS reads done
    }
}

// ---------------------------------------------------------------------------
// Kernel C: conv2 1x1 64->3 + bias, f16 in, f32 NCHW out
// ---------------------------------------------------------------------------
__global__ __launch_bounds__(256) void kC_conv2(
    const _Float16* __restrict__ h,
    const float* __restrict__ w,      // [3][64]
    const float* __restrict__ b,      // [3]
    float* __restrict__ out)          // [3][512][512]
{
    int p = blockIdx.x * 256 + threadIdx.x;
    const _Float16* hp = h + ((size_t)p << 6);
    float s0 = b[0], s1 = b[1], s2 = b[2];
#pragma unroll
    for (int q = 0; q < 8; ++q) {
        v8h vv = *(const v8h*)(hp + q * 8);
#pragma unroll
        for (int t = 0; t < 8; ++t) {
            int ic = q * 8 + t;
            float v = (float)vv[t];
            s0 += v * w[ic];
            s1 += v * w[64 + ic];
            s2 += v * w[128 + ic];
        }
    }
    out[p]              = s0;
    out[NPIX + p]       = s1;
    out[2 * NPIX + p]   = s2;
}

// ---------------------------------------------------------------------------
extern "C" void kernel_launch(void* const* d_in, const int* in_sizes, int n_in,
                              void* d_out, int out_size, void* d_ws, size_t ws_size,
                              hipStream_t stream)
{
    const float* x       = (const float*)d_in[0];
    // d_in[1] = out_size scalar (512), hardcoded
    const float* conv1_w = (const float*)d_in[2];
    const float* conv1_b = (const float*)d_in[3];
    const float* conv2_w = (const float*)d_in[4];
    const float* conv2_b = (const float*)d_in[5];
    const float* mlp_w1  = (const float*)d_in[6];
    const float* mlp_b1  = (const float*)d_in[7];
    const float* mlp_w2  = (const float*)d_in[8];
    const float* mlp_b2  = (const float*)d_in[9];

    _Float16* feat  = (_Float16*)d_ws;                       // 512*512*64 f16 = 32MB
    _Float16* c1out = feat + (size_t)NPIX * 64;              // 32MB
    _Float16* bpack = c1out + (size_t)NPIX * 64;             // 72KB

    kA_feat<<<NPIX / 256, 256, 0, stream>>>(x, mlp_w1, mlp_b1, mlp_w2, mlp_b2, feat);
    kB0_pack<<<(KCHUNKS * 2 * 64 * 16 + 255) / 256, 256, 0, stream>>>(conv1_w, bpack);
    kB_conv1<<<(512 / 4) * 16, 256, 0, stream>>>(feat, bpack, conv1_b, c1out);
    kC_conv2<<<NPIX / 256, 256, 0, stream>>>(c1out, conv2_w, conv2_b, (float*)d_out);
}
